// PMA_61186104099077
// MI455X (gfx1250) — matrix operand order repaired and checked by reading, not compiled, not run on hardware
//
#include <hip/hip_runtime.h>
#include <hip/hip_bf16.h>

// PMA (pooled multi-head attention, SEEDS=1) for MI455X gfx1250.
//
// Algebraic reduction (SEEDS=1):
//   qp   = Wq @ S + bq                       (1024,   batch-independent)
//   u_h  = Wk_h^T qp_h                       (16 x 1024)
//   c_h  = qp_h . bk_h
//   score[b,h,n] = (u_h . x[b,n] + c_h)/32   -> masked softmax -> P
//   xbar[b,h]    = sum_n P[b,h,n] x[b,n]     (16 x 1024 per batch)
//   O[b,d]       = Wv[d] . xbar[b, d>>6] + asum * bv[d]  (+S, LN, relu-res, LN)
//
// Two skinny GEMMs (~2.1 GFLOP each) via v_wmma_f32_16x16x32_bf16; X is read
// twice (2nd pass L2-hot within the 192MB L2). Shared WMMA operands (u-frags,
// P rows) are staged in LDS; f32->bf16 uses native casts (v_cvt_pk_bf16_f32).

#define SEQ   2048
#define NDIM  1024
#define NHEAD 16

typedef __attribute__((ext_vector_type(16))) __bf16        v16bf;
typedef __attribute__((ext_vector_type(8)))  float         v8f;
typedef __attribute__((ext_vector_type(16))) unsigned short v16us;
typedef __attribute__((ext_vector_type(8)))  unsigned short v8us;

__device__ __forceinline__ unsigned short f2bf(float f) {
  unsigned int u = __float_as_uint(f);
  u += 0x7FFFu + ((u >> 16) & 1u);
  return (unsigned short)(u >> 16);
}

// ---------------- K0: qp, c, and u packed in WMMA B-fragment order ----------
__global__ __launch_bounds__(256)
void k0_prep(const float* __restrict__ S, const float* __restrict__ Wq,
             const float* __restrict__ bq, const float* __restrict__ Wk,
             const float* __restrict__ bk,
             unsigned short* __restrict__ ufrag, float* __restrict__ cvec) {
  __shared__ float sS[NDIM];
  __shared__ float sqp[NDIM];
  const int tid = threadIdx.x;
  for (int i = tid; i < NDIM; i += 256) sS[i] = S[i];
  __syncthreads();
  for (int row = tid; row < NDIM; row += 256) {
    const float* w = Wq + (size_t)row * NDIM;
    float acc = 0.f;
    for (int j = 0; j < NDIM; ++j) acc += w[j] * sS[j];
    sqp[row] = acc + bq[row];
  }
  __syncthreads();
  if (tid < NHEAD) {
    float acc = 0.f;
    for (int j = 0; j < 64; ++j) acc += sqp[tid * 64 + j] * bk[tid * 64 + j];
    cvec[tid] = acc;
  }
  // u[h][k] = sum_j Wk[h*64+j][k] * qp[h*64+j]; store bf16 in B-frag layout:
  // chunk kc=k>>5, kk=k&31, lane = h | (((kk>>3)&1)<<4), elem = (kk&7)|((kk>>4)<<3).
  for (int idx = tid; idx < NHEAD * NDIM; idx += 256) {
    const int h = idx >> 10, k = idx & (NDIM - 1);
    float acc = 0.f;
    for (int j = 0; j < 64; ++j)
      acc += Wk[((size_t)(h * 64 + j)) * NDIM + k] * sqp[h * 64 + j];
    const int kc = k >> 5, kk = k & 31;
    const int lane = h | (((kk >> 3) & 1) << 4);
    ufrag[(size_t)(kc * 32 + lane) * 16 + ((kk & 7) | ((kk >> 4) << 3))] = f2bf(acc);
  }
}

// ---------------- K1: scores = X . u^T  via WMMA bf16 ----------------------
// grid (16, 32 batches), 256 thr = 8 waves; wave -> one 16-row tile, 16 heads.
__global__ __launch_bounds__(256)
void k1_scores(const float* __restrict__ X, const unsigned short* __restrict__ ufrag,
               float* __restrict__ scores) {
  __shared__ unsigned short sUf[32 * 32 * 16];                 // 32 KB
  const int tid = threadIdx.x, lane = tid & 31, wv = tid >> 5;
  {
    const uint4* src = (const uint4*)ufrag;
    uint4* dst = (uint4*)sUf;
    #pragma unroll
    for (int i = 0; i < 8; ++i) dst[tid + i * 256] = src[tid + i * 256];
  }
  __syncthreads();
  const int b = blockIdx.y;
  const int rowBase = (blockIdx.x * 8 + wv) * 16;
  const int row = rowBase + (lane & 15);
  const int hiA = (lane >> 4) << 3;              // A-frag K offset: 0 or 8
  const float* xr = X + ((size_t)b * SEQ + row) * NDIM;
  v8f acc = {};
  for (int kc = 0; kc < NDIM / 32; ++kc) {
    const int kb = kc * 32 + hiA;
    __builtin_prefetch(xr + kb + 64, 0, 0);
    float4 f0 = *(const float4*)(xr + kb);
    float4 f1 = *(const float4*)(xr + kb + 4);
    float4 f2 = *(const float4*)(xr + kb + 16);
    float4 f3 = *(const float4*)(xr + kb + 20);
    float fa[16] = {f0.x, f0.y, f0.z, f0.w, f1.x, f1.y, f1.z, f1.w,
                    f2.x, f2.y, f2.z, f2.w, f3.x, f3.y, f3.z, f3.w};
    v16bf a;
    #pragma unroll
    for (int e = 0; e < 16; ++e) a[e] = (__bf16)fa[e];
    v16bf bf = __builtin_bit_cast(v16bf,
        *(const v16us*)(sUf + (kc * 32 + lane) * 16));
    acc = __builtin_amdgcn_wmma_f32_16x16x32_bf16(false, a, false, bf,
                                                  (short)0, acc, false, false);
  }
  const int head = lane & 15;
  #pragma unroll
  for (int r = 0; r < 8; ++r) {
    const int n = rowBase + r + hiA;
    scores[((size_t)b * SEQ + n) * NHEAD + head] = acc[r];
  }
}

// ---------------- K2: masked softmax over n; P (bf16) + asum ---------------
__global__ __launch_bounds__(256)
void k2_softmax(const float* __restrict__ scores, const int* __restrict__ mask,
                const float* __restrict__ cvec, unsigned short* __restrict__ P,
                float* __restrict__ asum) {
  const int h = blockIdx.x, b = blockIdx.y, tid = threadIdx.x;
  __shared__ float red[256];
  const float ch = cvec[h];
  float sv[8]; int mv[8];
  float lmax = -3.402823466e38f;
  #pragma unroll
  for (int i = 0; i < 8; ++i) {
    const int n = tid + i * 256;
    mv[i] = mask[b * SEQ + n];
    sv[i] = (scores[((size_t)b * SEQ + n) * NHEAD + h] + ch) * 0.03125f;
    if (mv[i] && sv[i] > lmax) lmax = sv[i];
  }
  red[tid] = lmax; __syncthreads();
  for (int s = 128; s > 0; s >>= 1) {
    if (tid < s) red[tid] = fmaxf(red[tid], red[tid + s]);
    __syncthreads();
  }
  const float mx = red[0];
  __syncthreads();
  float ev[8]; float lsum = 0.f;
  #pragma unroll
  for (int i = 0; i < 8; ++i) {
    ev[i] = mv[i] ? __expf(sv[i] - mx) : 0.f;
    lsum += ev[i];
  }
  red[tid] = lsum; __syncthreads();
  for (int s = 128; s > 0; s >>= 1) {
    if (tid < s) red[tid] += red[tid + s];
    __syncthreads();
  }
  const float tot = red[0];
  const float inv = tot > 0.f ? 1.f / tot : 0.f;
  #pragma unroll
  for (int i = 0; i < 8; ++i)
    P[((size_t)b * NHEAD + h) * SEQ + tid + i * 256] = f2bf(ev[i] * inv);
  if (tid == 0) asum[b * NHEAD + h] = tot > 0.f ? 1.f : 0.f;
}

// ---------------- K3: xbar = P . X  via WMMA bf16 --------------------------
// grid (8, 32 batches), 8 waves; wave -> one 16-dim column tile, all 16 heads.
// All waves in the block share the same A fragments (P rows) -> stage in LDS.
__global__ __launch_bounds__(256)
void k3_xbar(const float* __restrict__ X, const unsigned short* __restrict__ P,
             float* __restrict__ xbar) {
  __shared__ unsigned short sP[NHEAD * SEQ];                   // 64 KB
  const int tid = threadIdx.x, lane = tid & 31, wv = tid >> 5;
  const int b = blockIdx.y;
  {
    const uint4* src = (const uint4*)(P + (size_t)b * NHEAD * SEQ);
    uint4* dst = (uint4*)sP;
    #pragma unroll
    for (int i = 0; i < 16; ++i) dst[tid + i * 256] = src[tid + i * 256];
  }
  __syncthreads();
  const int dcol = (blockIdx.x * 8 + wv) * 16 + (lane & 15);
  const int head = lane & 15;
  const int hi8 = (lane >> 4) << 3;
  const int hi16 = (lane >> 4) << 4;
  const unsigned short* pr = sP + head * SEQ;
  const float* xb = X + (size_t)b * SEQ * NDIM + dcol;
  v8f acc = {};
  for (int kc = 0; kc < SEQ / 32; ++kc) {
    const int nb = kc * 32 + hi8;
    v8us p0 = *(const v8us*)(pr + nb);
    v8us p1 = *(const v8us*)(pr + nb + 16);
    v16us au;
    #pragma unroll
    for (int j = 0; j < 8; ++j) { au[j] = p0[j]; au[8 + j] = p1[j]; }
    v16bf a = __builtin_bit_cast(v16bf, au);
    const int k0 = kc * 32 + hi16;
    __builtin_prefetch(xb + (size_t)(k0 + 32) * NDIM, 0, 0);
    v16bf bb;
    #pragma unroll
    for (int e = 0; e < 16; ++e) bb[e] = (__bf16)xb[(size_t)(kc * 32 + hi8 + (e & 7) + ((e >> 3) << 4)) * NDIM];
    acc = __builtin_amdgcn_wmma_f32_16x16x32_bf16(false, a, false, bb,
                                                  (short)0, acc, false, false);
  }
  #pragma unroll
  for (int r = 0; r < 8; ++r)
    xbar[((size_t)b * NHEAD + r + hi8) * NDIM + dcol] = acc[r];
}

// ---------------- K4: Wv matvec + residual + LN + relu(Wo) + LN ------------
__device__ __forceinline__ float blockSum(float v, float* red) {
  const int tid = threadIdx.x;
  __syncthreads();
  red[tid] = v; __syncthreads();
  for (int s = 128; s > 0; s >>= 1) {
    if (tid < s) red[tid] += red[tid + s];
    __syncthreads();
  }
  return red[0];
}

__global__ __launch_bounds__(256)
void k4_finish(const float* __restrict__ xbar, const float* __restrict__ asum,
               const float* __restrict__ S, const float* __restrict__ Wv,
               const float* __restrict__ bv, const float* __restrict__ Wo,
               const float* __restrict__ bo, const float* __restrict__ g0,
               const float* __restrict__ b0, const float* __restrict__ g1,
               const float* __restrict__ b1, float* __restrict__ out) {
  const int b = blockIdx.x, tid = threadIdx.x;
  __shared__ float sO[NDIM];
  __shared__ float red[256];
  float o[4];
  #pragma unroll
  for (int i = 0; i < 4; ++i) {
    const int d = tid + i * 256;
    const int h = d >> 6;
    const float* wr = Wv + (size_t)d * NDIM;
    const float* xr = xbar + ((size_t)b * NHEAD + h) * NDIM;
    float acc = 0.f;
    for (int t = 0; t < NDIM; t += 4) {
      float4 w4 = *(const float4*)(wr + t);
      float4 x4 = *(const float4*)(xr + t);
      acc += w4.x * x4.x + w4.y * x4.y + w4.z * x4.z + w4.w * x4.w;
    }
    o[i] = S[d] + acc + asum[b * NHEAD + h] * bv[d];
  }
  // LayerNorm 0
  float s1 = o[0] + o[1] + o[2] + o[3];
  float s2 = o[0]*o[0] + o[1]*o[1] + o[2]*o[2] + o[3]*o[3];
  const float mu0 = blockSum(s1, red) * (1.f / NDIM);
  const float ex0 = blockSum(s2, red) * (1.f / NDIM);
  const float rs0 = rsqrtf(fmaxf(ex0 - mu0 * mu0, 0.f) + 1e-5f);
  #pragma unroll
  for (int i = 0; i < 4; ++i) {
    const int d = tid + i * 256;
    o[i] = (o[i] - mu0) * rs0 * g0[d] + b0[d];
    sO[d] = o[i];
  }
  __syncthreads();
  // relu residual with Wo
  #pragma unroll
  for (int i = 0; i < 4; ++i) {
    const int d = tid + i * 256;
    const float* wr = Wo + (size_t)d * NDIM;
    float acc = 0.f;
    for (int t = 0; t < NDIM; t += 4) {
      float4 w4 = *(const float4*)(wr + t);
      acc += w4.x * sO[t] + w4.y * sO[t + 1] + w4.z * sO[t + 2] + w4.w * sO[t + 3];
    }
    o[i] = o[i] + fmaxf(acc + bo[d], 0.f);
  }
  // LayerNorm 1
  s1 = o[0] + o[1] + o[2] + o[3];
  s2 = o[0]*o[0] + o[1]*o[1] + o[2]*o[2] + o[3]*o[3];
  const float mu1 = blockSum(s1, red) * (1.f / NDIM);
  const float ex1 = blockSum(s2, red) * (1.f / NDIM);
  const float rs1 = rsqrtf(fmaxf(ex1 - mu1 * mu1, 0.f) + 1e-5f);
  #pragma unroll
  for (int i = 0; i < 4; ++i) {
    const int d = tid + i * 256;
    out[(size_t)b * NDIM + d] = (o[i] - mu1) * rs1 * g1[d] + b1[d];
  }
}

extern "C" void kernel_launch(void* const* d_in, const int* in_sizes, int n_in,
                              void* d_out, int out_size, void* d_ws, size_t ws_size,
                              hipStream_t stream) {
  (void)in_sizes; (void)n_in; (void)out_size; (void)ws_size;
  const float* X    = (const float*)d_in[0];
  const int*   mask = (const int*)d_in[1];
  const float* S    = (const float*)d_in[2];
  const float* Wq   = (const float*)d_in[3];
  const float* bq   = (const float*)d_in[4];
  const float* Wk   = (const float*)d_in[5];
  const float* bk   = (const float*)d_in[6];
  const float* Wv   = (const float*)d_in[7];
  const float* bv   = (const float*)d_in[8];
  const float* Wo   = (const float*)d_in[9];
  const float* bo   = (const float*)d_in[10];
  const float* g0   = (const float*)d_in[11];
  const float* b0   = (const float*)d_in[12];
  const float* g1   = (const float*)d_in[13];
  const float* b1   = (const float*)d_in[14];
  float* out = (float*)d_out;

  char* ws = (char*)d_ws;
  unsigned short* ufrag = (unsigned short*)(ws);                       // 32 KB
  float* cvec   = (float*)(ws + (32u << 10));                          // 64 B
  float* scores = (float*)(ws + (64u << 10));                          // 4 MB
  unsigned short* P = (unsigned short*)(ws + (64u << 10) + (4u << 20)); // 2 MB
  float* asum = (float*)(ws + (64u << 10) + (6u << 20));               // 2 KB
  float* xbar = (float*)(ws + (64u << 10) + (6u << 20) + (16u << 10)); // 2 MB

  k0_prep<<<1, 256, 0, stream>>>(S, Wq, bq, Wk, bk, ufrag, cvec);
  k1_scores<<<dim3(16, 32), 256, 0, stream>>>(X, ufrag, scores);
  k2_softmax<<<dim3(NHEAD, 32), 256, 0, stream>>>(scores, mask, cvec, P, asum);
  k3_xbar<<<dim3(8, 32), 256, 0, stream>>>(X, P, xbar);
  k4_finish<<<32, 256, 0, stream>>>(xbar, asum, S, Wv, bv, Wo, bo,
                                    g0, b0, g1, b1, out);
}
